// LSTM_82240033784242
// MI455X (gfx1250) — matrix operand order repaired and checked by reading, not compiled, not run on hardware
//
#include <hip/hip_runtime.h>

// ---------------------------------------------------------------------------
// Fused LSTM for MI455X (gfx1250, wave32, WMMA bf16 16x16x32, f32 accum).
//
// Round-4 changes:
//  * K loop split into x-phase / boundary / h-phase: no per-iteration branch,
//    no implicit-def register diamonds.
//  * steady loops #pragma unroll 3 (15 = 3x5): unroller rotates pipelined load
//    destination registers -> no WAR waits before issuing next loads.
//  * hoisted per-thread base pointers so pipelined loads become single-base +
//    immediate-offset global_load_b128 / ds_load_b128.
// ---------------------------------------------------------------------------

#define N_DIM   64
#define T_DIM   512
#define D_DIM   512
#define H_DIM   512
#define FOURH   2048
#define KCH     32
#define NKB_X   16               // x-projection K chunks
#define NKB     32               // total K chunks (x + h)
#define NTILES  (FOURH / 16)     // 128
#define NT_GATE (H_DIM / 16)     // 32
#define NWG     32
#define XPSTEP  (NKB_X * 4 * 32 * 16)   // xpack halves per timestep (32768)
#define BSTRIDE (4 * 32 * 16)           // Wlds halves per K-chunk (2048)

typedef __attribute__((ext_vector_type(16))) __bf16 v16bf;
typedef __attribute__((ext_vector_type(8)))  __bf16 v8bf;
typedef __attribute__((ext_vector_type(4)))  __bf16 v4bf;
typedef __attribute__((ext_vector_type(8)))  float  v8f;
typedef __attribute__((ext_vector_type(4)))  float  v4f;

// Workspace layout (bytes)
#define WS_WPACK 0u          // 4 MiB weights, B-fragment order
#define WS_XPACK 4194304u    // 32 MiB x, A-fragment order
#define WS_HBF   37748736u   // 64*512 bf16 hidden state
#define WS_CNT   37814272u   // grid-barrier monotonic counter

__launch_bounds__(256)
__global__ void lstm_pack_w(const float* __restrict__ Wx,
                            const float* __restrict__ Wh,
                            __bf16* __restrict__ Wpack) {
  const int total = NKB * NTILES * 32 * 16;
  for (int p = blockIdx.x * blockDim.x + threadIdx.x; p < total;
       p += gridDim.x * blockDim.x) {
    int e     = p & 15;
    int lane  = (p >> 4) & 31;
    int ntile = (p >> 9) & 127;
    int kb    = p >> 16;
    int k = kb * KCH + ((lane >> 4) << 3) + (e & 7) + ((e >> 3) << 4);
    int n = ntile * 16 + (lane & 15);
    float v = (k < D_DIM) ? Wx[k * FOURH + n] : Wh[(k - D_DIM) * FOURH + n];
    Wpack[p] = (__bf16)v;
  }
}

__launch_bounds__(256)
__global__ void lstm_pack_x(const float* __restrict__ x,
                            __bf16* __restrict__ xpack) {
  const int total = T_DIM * XPSTEP;  // 16,777,216
  for (int p = blockIdx.x * blockDim.x + threadIdx.x; p < total;
       p += gridDim.x * blockDim.x) {
    int e    = p & 15;
    int lane = (p >> 4) & 31;
    int rt   = (p >> 9) & 3;
    int kb   = (p >> 11) & 15;
    int t    = p >> 15;
    int row = rt * 16 + (lane & 15);
    int k   = kb * KCH + ((lane >> 4) << 3) + (e & 7) + ((e >> 3) << 4);
    xpack[p] = (__bf16)x[(row * T_DIM + t) * D_DIM + k];
  }
}

__launch_bounds__(256)
__global__ void lstm_state_init(const float* __restrict__ h0,
                                __bf16* __restrict__ h_st,
                                unsigned* __restrict__ cnt) {
  int tid = blockIdx.x * blockDim.x + threadIdx.x;
  if (tid < N_DIM * H_DIM) h_st[tid] = (__bf16)h0[tid];
  if (tid == 0) *cnt = 0u;
}

__device__ __forceinline__ float fast_sigmoid(float x) {
  return 1.0f / (1.0f + __expf(-x));
}
__device__ __forceinline__ float fast_tanh(float x) {
  float e = __expf(2.0f * x);
  return (e - 1.0f) / (e + 1.0f);
}

#define WMMA_BF16(A, B, C)                                                   \
  __builtin_amdgcn_wmma_f32_16x16x32_bf16(false, (A), false, (B), (short)0,  \
                                          (C), false, false)

__launch_bounds__(256, 1)
__global__ void lstm_scan(const __bf16* __restrict__ xpack,
                          const float* __restrict__ bias,
                          const __bf16* __restrict__ Wpack,
                          __bf16* __restrict__ h_st,
                          unsigned* __restrict__ cnt,
                          float* __restrict__ out) {
  __shared__ __bf16 Wlds[NKB * BSTRIDE];  // [kb][gate][lane][e] = 128 KB
  __shared__ float  AT[N_DIM * 64];       // gate pre-activations, 16 KB

  const int tid   = threadIdx.x;
  const int wg    = blockIdx.x;
  const int w     = tid >> 5;
  const int lane  = tid & 31;
  const int rtile = w & 3;
  const int g0    = (w >> 2) * 2;

  const int arow = rtile * 16 + (lane & 15);
  const int koff = (lane >> 4) * 8;

  // ---- preload this WG's weight slice (4 gates x 32 kb) into LDS ----
  for (int c = tid; c < NKB * 4 * 32; c += 256) {
    int kb = c >> 7;
    int G  = (c >> 5) & 3;
    int ln = c & 31;
    *(v16bf*)&Wlds[c * 16] =
        *(const v16bf*)&Wpack[((kb * NTILES + G * NT_GATE + wg) * 32 + ln) * 16];
  }
  __syncthreads();

  // ---- per-thread fixed (n, 4 consecutive hcol) ownership ----
  const int p4    = tid * 4;
  const int en    = p4 >> 4;       // batch row 0..63
  const int el16  = p4 & 15;       // col offset within WG's 16 (multiple of 4)
  const int ehcol = wg * 16 + el16;

  const v4f bi = *(const v4f*)&bias[ehcol];
  const v4f bf = *(const v4f*)&bias[H_DIM + ehcol];
  const v4f bo = *(const v4f*)&bias[2 * H_DIM + ehcol];
  const v4f bg = *(const v4f*)&bias[3 * H_DIM + ehcol];

  v4f c_reg = {0.f, 0.f, 0.f, 0.f};

  // hoisted per-thread bases -> immediate-offset loads in the unrolled loops
  const __bf16* lb0  = &Wlds[(g0 * 32 + lane) * 16];        // B, gate g0
  const __bf16* lb1  = &Wlds[((g0 + 1) * 32 + lane) * 16];  // B, gate g0+1
  const __bf16* hrow = h_st + arow * H_DIM + koff;          // A (h part)
  const __bf16* xa0  = xpack + (rtile * 32 + lane) * 16;    // A (x part), t=0

  for (int t = 0; t < T_DIM; ++t) {
    const __bf16* xa = xa0 + t * XPSTEP;

    v8f acc0 = {};
    v8f acc1 = {};

    // ---- prologue: fragments for chunk 0 ----
    v16bf a_c  = *(const v16bf*)&xa[0];
    v16bf b0_c = *(const v16bf*)&lb0[0];
    v16bf b1_c = *(const v16bf*)&lb1[0];

    // ---- x phase: chunks 0..14, preload chunk kb+1 from xpack ----
#pragma unroll 3
    for (int kb = 0; kb < NKB_X - 1; ++kb) {
      v16bf a_n  = *(const v16bf*)&xa[(kb + 1) * (4 * 32 * 16)];
      v16bf b0_n = *(const v16bf*)&lb0[(kb + 1) * BSTRIDE];
      v16bf b1_n = *(const v16bf*)&lb1[(kb + 1) * BSTRIDE];
      acc0 = WMMA_BF16(a_c, b0_c, acc0);
      acc1 = WMMA_BF16(a_c, b1_c, acc1);
      a_c = a_n; b0_c = b0_n; b1_c = b1_n;
    }

    // ---- boundary: compute chunk 15, preload first h chunk ----
    {
      v8bf alo = *(const v8bf*)&hrow[0];
      v8bf ahi = *(const v8bf*)&hrow[16];
      v16bf a_n = __builtin_shufflevector(alo, ahi,
          0, 1, 2, 3, 4, 5, 6, 7, 8, 9, 10, 11, 12, 13, 14, 15);
      v16bf b0_n = *(const v16bf*)&lb0[NKB_X * BSTRIDE];
      v16bf b1_n = *(const v16bf*)&lb1[NKB_X * BSTRIDE];
      acc0 = WMMA_BF16(a_c, b0_c, acc0);
      acc1 = WMMA_BF16(a_c, b1_c, acc1);
      a_c = a_n; b0_c = b0_n; b1_c = b1_n;
    }

    // ---- h phase: chunks 16..30, preload next h chunk ----
#pragma unroll 3
    for (int kk = 0; kk < NKB_X - 1; ++kk) {
      v8bf alo = *(const v8bf*)&hrow[(kk + 1) * KCH];
      v8bf ahi = *(const v8bf*)&hrow[(kk + 1) * KCH + 16];
      v16bf a_n = __builtin_shufflevector(alo, ahi,
          0, 1, 2, 3, 4, 5, 6, 7, 8, 9, 10, 11, 12, 13, 14, 15);
      v16bf b0_n = *(const v16bf*)&lb0[(kk + NKB_X + 1) * BSTRIDE];
      v16bf b1_n = *(const v16bf*)&lb1[(kk + NKB_X + 1) * BSTRIDE];
      acc0 = WMMA_BF16(a_c, b0_c, acc0);
      acc1 = WMMA_BF16(a_c, b1_c, acc1);
      a_c = a_n; b0_c = b0_n; b1_c = b1_n;
    }

    // ---- epilogue: chunk 31 ----
    acc0 = WMMA_BF16(a_c, b0_c, acc0);
    acc1 = WMMA_BF16(a_c, b1_c, acc1);

    // ---- stream next timestep's x fragments toward L2 (t=511 over-prefetch
    //      lands in the mapped h_st region; harmless) ----
    {
      const __bf16* xp_n = xa + XPSTEP;
#pragma unroll
      for (int kb = 0; kb < NKB_X; ++kb)
        __builtin_prefetch(&xp_n[kb * (4 * 32 * 16)], 0, 1);
    }

    // ---- spill accumulators (C layout: VGPR r -> M = r + 8*(lane>>4)) ----
#pragma unroll
    for (int r = 0; r < 8; ++r) {
      int m = r + (lane >> 4) * 8;
      int n = lane & 15;
      AT[(rtile * 16 + m) * 64 + g0 * 16 + n]       = acc0[r];
      AT[(rtile * 16 + m) * 64 + (g0 + 1) * 16 + n] = acc1[r];
    }
    __syncthreads();

    // ---- vectorized WG-local gate math; c lives in registers ----
    {
      v4f ai = *(const v4f*)&AT[en * 64 +  0 + el16] + bi;
      v4f af = *(const v4f*)&AT[en * 64 + 16 + el16] + bf;
      v4f ao = *(const v4f*)&AT[en * 64 + 32 + el16] + bo;
      v4f ag = *(const v4f*)&AT[en * 64 + 48 + el16] + bg;

      v4f hn;
      v4bf hb;
#pragma unroll
      for (int j = 0; j < 4; ++j) {
        float ig = fast_sigmoid(ai[j]);
        float fg = fast_sigmoid(af[j]);
        float og = fast_sigmoid(ao[j]);
        float gg = fast_tanh(ag[j]);
        float cn = fg * c_reg[j] + ig * gg;
        c_reg[j] = cn;
        float h = og * fast_tanh(cn);
        hn[j] = h;
        hb[j] = (__bf16)h;
      }
      *(v4f*)&out[(en * T_DIM + t) * H_DIM + ehcol] = hn;
      *(v4bf*)&h_st[en * H_DIM + ehcol] = hb;
    }

    // ---- grid barrier: publish h_st before anyone starts step t+1 ----
    __syncthreads();
    if (tid == 0) {
      __threadfence();
      __hip_atomic_fetch_add(cnt, 1u, __ATOMIC_RELEASE, __HIP_MEMORY_SCOPE_AGENT);
      unsigned target = (unsigned)(NWG * (t + 1));
      while (__hip_atomic_load(cnt, __ATOMIC_ACQUIRE, __HIP_MEMORY_SCOPE_AGENT) < target) {
        __builtin_amdgcn_s_sleep(2);
      }
    }
    __syncthreads();
  }
}

extern "C" void kernel_launch(void* const* d_in, const int* in_sizes, int n_in,
                              void* d_out, int out_size, void* d_ws, size_t ws_size,
                              hipStream_t stream) {
  const float* x  = (const float*)d_in[0];  // (64, 512, 512)
  const float* h0 = (const float*)d_in[1];  // (64, 512)
  const float* Wx = (const float*)d_in[2];  // (512, 2048)
  const float* Wh = (const float*)d_in[3];  // (512, 2048)
  const float* b  = (const float*)d_in[4];  // (2048,)
  float* out = (float*)d_out;               // (64, 512, 512)

  char* ws = (char*)d_ws;
  __bf16*   Wpack = (__bf16*)(ws + WS_WPACK);
  __bf16*   xpack = (__bf16*)(ws + WS_XPACK);
  __bf16*   h_st  = (__bf16*)(ws + WS_HBF);
  unsigned* cnt   = (unsigned*)(ws + WS_CNT);

  lstm_pack_w<<<1024, 256, 0, stream>>>(Wx, Wh, Wpack);
  lstm_pack_x<<<8192, 256, 0, stream>>>(x, xpack);
  lstm_state_init<<<128, 256, 0, stream>>>(h0, h_st, cnt);
  lstm_scan<<<NWG, 256, 0, stream>>>(xpack, b, Wpack, h_st, cnt, out);
}
